// CasualSelfAttention_62319975465123
// MI455X (gfx1250) — matrix operand order repaired and checked by reading, not compiled, not run on hardware
//
#include <hip/hip_runtime.h>
#include <hip/hip_bf16.h>

// B=4, T=2048, C=1024, H=16, HS=64
#define TB 4
#define TT 2048
#define TC 1024
#define TH 16
#define THS 64

typedef __attribute__((ext_vector_type(16))) __bf16          bf16x16;
typedef __attribute__((ext_vector_type(8)))  float           f32x8;
typedef __attribute__((ext_vector_type(4)))  float           f32x4;
typedef __attribute__((ext_vector_type(4)))  unsigned short  u16x4;
typedef __attribute__((ext_vector_type(8)))  unsigned short  u16x8;
typedef __attribute__((ext_vector_type(16))) unsigned short  u16x16;

__device__ __forceinline__ unsigned short f2bf(float f) {
    unsigned u = __builtin_bit_cast(unsigned, f);
    u += 0x7FFFu + ((u >> 16) & 1u);          // round-to-nearest-even
    return (unsigned short)(u >> 16);
}

__device__ __forceinline__ bf16x16 mkfrag(u16x8 lo, u16x8 hi) {
    u16x16 t = __builtin_shufflevector(lo, hi, 0, 1, 2, 3, 4, 5, 6, 7,
                                       8, 9, 10, 11, 12, 13, 14, 15);
    return __builtin_bit_cast(bf16x16, t);
}

__device__ __forceinline__ f32x8 zero8() {
    f32x8 v;
#pragma unroll
    for (int i = 0; i < 8; ++i) v[i] = 0.0f;
    return v;
}

__device__ __forceinline__ f32x8 wmma_bf16(bf16x16 a, bf16x16 b, f32x8 c) {
    return __builtin_amdgcn_wmma_f32_16x16x32_bf16(false, a, false, b,
                                                   (short)0, c, false, false);
}

// ---------------------------------------------------------------------------
// Prologue A: elementwise f32 -> bf16 (x -> xb), vectorized 4-wide.
// ---------------------------------------------------------------------------
__global__ void __launch_bounds__(256)
cvt_bf16_kernel(const float* __restrict__ in, unsigned short* __restrict__ out,
                int n4) {
    int i = blockIdx.x * 256 + threadIdx.x;
    if (i < n4) {
        f32x4 v = ((const f32x4*)in)[i];
        u16x4 o;
#pragma unroll
        for (int j = 0; j < 4; ++j) o[j] = f2bf(v[j]);
        ((u16x4*)out)[i] = o;
    }
}

// ---------------------------------------------------------------------------
// Prologue B: transpose + convert: in f32 [R][Ccols] -> out bf16 [Ccols][R].
// 32x32 tiles through LDS, coalesced on both sides.
// ---------------------------------------------------------------------------
__global__ void __launch_bounds__(256)
transpose_bf16_kernel(const float* __restrict__ in,
                      unsigned short* __restrict__ out, int R, int Ccols) {
    __shared__ float tile[32][33];
    const int bx = blockIdx.x * 32;           // col base (input)
    const int by = blockIdx.y * 32;           // row base (input)
    const int tx = threadIdx.x & 31;
    const int ty = threadIdx.x >> 5;          // 0..7
#pragma unroll
    for (int i = 0; i < 4; ++i)
        tile[ty + 8 * i][tx] = in[(size_t)(by + ty + 8 * i) * Ccols + bx + tx];
    __syncthreads();
#pragma unroll
    for (int i = 0; i < 4; ++i)
        out[(size_t)(bx + ty + 8 * i) * R + by + tx] = f2bf(tile[tx][ty + 8 * i]);
}

// ---------------------------------------------------------------------------
// Kernel 1: qkv = xb @ Wqkv + b, RoPE on q,k.  A,B already bf16 in global:
// main loop is pure global_load_b128 -> wmma.  Block = 4 waves; block tile
// 128(M) x 64(N); each wave 32 x 64 (2 A-frags x 4 B-frags = 8 wmma / K-step).
// Outputs: Qb,Kb bf16 [B,H,T,HS];  Vt bf16 [B,H,HS,T].
// ---------------------------------------------------------------------------
__global__ void __launch_bounds__(128)
qkv_rope_kernel(const unsigned short* __restrict__ xb,
                const unsigned short* __restrict__ Wqt,   // [3C][C] = W^T bf16
                const float* __restrict__ bqkv,
                unsigned short* __restrict__ Qb, unsigned short* __restrict__ Kb,
                unsigned short* __restrict__ Vt) {
    __shared__ float ldsC[4][16][68];

    const int tid  = threadIdx.x;
    const int lane = tid & 31;
    const int wave = tid >> 5;
    const int hv   = lane >> 4;
    const int l16  = lane & 15;

    const int m0 = blockIdx.x * 128;
    const int n0 = blockIdx.y * 64;

    f32x8 acc[2][4];
#pragma unroll
    for (int g = 0; g < 2; ++g)
#pragma unroll
        for (int c = 0; c < 4; ++c) acc[g][c] = zero8();

    const unsigned short* arow0 = xb + (size_t)(m0 + wave * 32 + l16) * TC;
    const unsigned short* arow1 = arow0 + 16 * TC;
    const unsigned short* bbase = Wqt + (size_t)(n0 + l16) * TC + hv * 16;

    for (int k0 = 0; k0 < TC; k0 += 32) {
        const u16x8* a0p = (const u16x8*)(arow0 + k0);
        const u16x8* a1p = (const u16x8*)(arow1 + k0);
        bf16x16 af0 = mkfrag(a0p[2 * hv], a0p[2 * hv + 1]);
        bf16x16 af1 = mkfrag(a1p[2 * hv], a1p[2 * hv + 1]);
        if (k0 + 32 < TC) {
            __builtin_prefetch(arow0 + k0 + 32, 0, 1);
            __builtin_prefetch(arow1 + k0 + 32, 0, 1);
        }
#pragma unroll
        for (int c = 0; c < 4; ++c) {
            const u16x8* bp = (const u16x8*)(bbase + (size_t)(16 * c) * TC + k0);
            bf16x16 bf = mkfrag(bp[0], bp[1]);
            acc[0][c] = wmma_bf16(af0, bf, acc[0][c]);
            acc[1][c] = wmma_bf16(af1, bf, acc[1][c]);
        }
    }

    const int mid = n0 >> 10;                 // 0=q,1=k,2=v
    const int h   = (n0 & 1023) >> 6;
    const float lth = 9.210340371976184f / 32.0f;   // ln(10000)/32

#pragma unroll
    for (int g = 0; g < 2; ++g) {
        __builtin_amdgcn_wave_barrier();      // order vs previous group's reads
#pragma unroll
        for (int c = 0; c < 4; ++c)
#pragma unroll
            for (int j = 0; j < 8; ++j)
                ldsC[wave][j + hv * 8][16 * c + l16] =
                    acc[g][c][j] + bqkv[n0 + 16 * c + l16];
        __builtin_amdgcn_wave_barrier();      // stores before cross-lane reads

        const int rbase = m0 + wave * 32 + g * 16;
#pragma unroll 4
        for (int t2 = 0; t2 < 32; ++t2) {
            int linear = t2 * 32 + lane;
            int row = linear >> 6;            // 0..15
            int col = linear & 63;            // 0..63
            int r = rbase + row;
            int b = r >> 11;
            int t = r & (TT - 1);
            if (mid == 2) {
                float v = ldsC[wave][row][col];
                Vt[(((size_t)b * TH + h) * THS + col) * TT + t] = f2bf(v);
            } else {
                int dp = (col < 32) ? col : (col - 32);
                float x1 = ldsC[wave][row][2 * dp];
                float x2 = ldsC[wave][row][2 * dp + 1];
                float theta = __expf(-(float)dp * lth);
                float sn, cs;
                __sincosf((float)t * theta, &sn, &cs);
                float o = (col < 32) ? (x1 * cs - x2 * sn) : (x1 * sn + x2 * cs);
                unsigned short* dst = (mid == 0) ? Qb : Kb;
                dst[(((size_t)b * TH + h) * TT + t) * THS + col] = f2bf(o);
            }
        }
    }
}

// ---------------------------------------------------------------------------
// Kernel 2: flash attention. 4 waves x 16 queries; key blocks of 32:
// S=QK^T (4 wmma), online softmax (half-wave shfl reductions), P relayout
// through wave-private LDS, P@V (4 wmma, V^T gives contiguous keys).
// ---------------------------------------------------------------------------
__global__ void __launch_bounds__(128)
attn_kernel(const unsigned short* __restrict__ Qb,
            const unsigned short* __restrict__ Kb,
            const unsigned short* __restrict__ Vt,
            unsigned short* __restrict__ attnout) {
    __shared__ unsigned short ldsP[4][16][40];

    const int tid  = threadIdx.x;
    const int lane = tid & 31;
    const int wave = tid >> 5;
    const int hv   = lane >> 4;
    const int l16  = lane & 15;

    const int bh = blockIdx.x;
    const int b  = bh >> 4, h = bh & 15;
    const int qw = blockIdx.y * 64 + wave * 16;

    const unsigned short* Qh = Qb + (size_t)bh * TT * THS;
    const unsigned short* Kh = Kb + (size_t)bh * TT * THS;
    const unsigned short* Vh = Vt + (size_t)bh * THS * TT;

    bf16x16 qfrag[2];
    {
        const u16x8* qp = (const u16x8*)(Qh + (size_t)(qw + l16) * THS);
        qfrag[0] = mkfrag(qp[2 * hv], qp[2 * hv + 1]);
        qfrag[1] = mkfrag(qp[2 * hv + 4], qp[2 * hv + 5]);
    }

    float mrun[8], lrun[8];
#pragma unroll
    for (int j = 0; j < 8; ++j) { mrun[j] = -3.0e38f; lrun[j] = 0.0f; }
    f32x8 acc[4] = {zero8(), zero8(), zero8(), zero8()};

    const float scale = 0.125f;
    const int kmax = qw + 16;

    for (int n0 = 0; n0 < kmax; n0 += 32) {
        f32x8 S[2] = {zero8(), zero8()};
#pragma unroll
        for (int hT = 0; hT < 2; ++hT) {
            int key = n0 + hT * 16 + l16;
            const u16x8* kp = (const u16x8*)(Kh + (size_t)key * THS);
            bf16x16 kf0 = mkfrag(kp[hv * 2], kp[hv * 2 + 1]);
            bf16x16 kf1 = mkfrag(kp[hv * 2 + 4], kp[hv * 2 + 5]);
            S[hT] = wmma_bf16(qfrag[0], kf0, S[hT]);
            S[hT] = wmma_bf16(qfrag[1], kf1, S[hT]);
        }

        float mnew[8];
#pragma unroll
        for (int j = 0; j < 8; ++j) {
            int q = qw + j + hv * 8;
#pragma unroll
            for (int hT = 0; hT < 2; ++hT) {
                int key = n0 + hT * 16 + l16;
                float s = S[hT][j] * scale;
                S[hT][j] = (key <= q) ? s : -3.0e38f;
            }
            float v = fmaxf(S[0][j], S[1][j]);
            v = fmaxf(v, __shfl_xor(v, 1, 32));
            v = fmaxf(v, __shfl_xor(v, 2, 32));
            v = fmaxf(v, __shfl_xor(v, 4, 32));
            v = fmaxf(v, __shfl_xor(v, 8, 32));
            mnew[j] = fmaxf(mrun[j], v);
        }

#pragma unroll
        for (int j = 0; j < 8; ++j) {
            float f = __expf(mrun[j] - mnew[j]);
            float rs = 0.0f;
#pragma unroll
            for (int hT = 0; hT < 2; ++hT) {
                float p = __expf(S[hT][j] - mnew[j]);
                S[hT][j] = p;
                rs += p;
            }
            rs += __shfl_xor(rs, 1, 32);
            rs += __shfl_xor(rs, 2, 32);
            rs += __shfl_xor(rs, 4, 32);
            rs += __shfl_xor(rs, 8, 32);
            lrun[j] = lrun[j] * f + rs;
            mrun[j] = mnew[j];
#pragma unroll
            for (int c = 0; c < 4; ++c) acc[c][j] *= f;
        }

        __builtin_amdgcn_wave_barrier();      // vs previous block's P reads
#pragma unroll
        for (int hT = 0; hT < 2; ++hT)
#pragma unroll
            for (int j = 0; j < 8; ++j)
                ldsP[wave][j + hv * 8][hT * 16 + l16] = f2bf(S[hT][j]);
        __builtin_amdgcn_wave_barrier();      // stores before cross-lane reads
        const u16x8* pp = (const u16x8*)&ldsP[wave][l16][0];
        bf16x16 pfrag = mkfrag(pp[2 * hv], pp[2 * hv + 1]);

#pragma unroll
        for (int c = 0; c < 4; ++c) {
            int hs = 16 * c + l16;
            const u16x8* vp = (const u16x8*)(Vh + (size_t)hs * TT + n0 + hv * 16);
            bf16x16 vfrag = mkfrag(vp[0], vp[1]);
            acc[c] = wmma_bf16(pfrag, vfrag, acc[c]);
        }
    }

#pragma unroll
    for (int c = 0; c < 4; ++c)
#pragma unroll
        for (int j = 0; j < 8; ++j) {
            int q  = qw + j + hv * 8;
            int hs = 16 * c + l16;
            float o = acc[c][j] / lrun[j];
            attnout[((size_t)b * TT + q) * TC + h * THS + hs] = f2bf(o);
        }
}

// ---------------------------------------------------------------------------
// Kernel 3: out = attn(bf16) @ W_o + b_o (f32).  Same pure-global WMMA loop,
// block tile 128x64, wave tile 32x64.
// ---------------------------------------------------------------------------
__global__ void __launch_bounds__(128)
oproj_kernel(const unsigned short* __restrict__ A,
             const unsigned short* __restrict__ Wot,   // [C][C] = W_o^T bf16
             const float* __restrict__ bo, float* __restrict__ out) {
    const int tid  = threadIdx.x;
    const int lane = tid & 31;
    const int wave = tid >> 5;
    const int hv   = lane >> 4;
    const int l16  = lane & 15;

    const int m0 = blockIdx.x * 128;
    const int n0 = blockIdx.y * 64;

    f32x8 acc[2][4];
#pragma unroll
    for (int g = 0; g < 2; ++g)
#pragma unroll
        for (int c = 0; c < 4; ++c) acc[g][c] = zero8();

    const unsigned short* arow0 = A + (size_t)(m0 + wave * 32 + l16) * TC;
    const unsigned short* arow1 = arow0 + 16 * TC;
    const unsigned short* bbase = Wot + (size_t)(n0 + l16) * TC + hv * 16;

    for (int k0 = 0; k0 < TC; k0 += 32) {
        const u16x8* a0p = (const u16x8*)(arow0 + k0);
        const u16x8* a1p = (const u16x8*)(arow1 + k0);
        bf16x16 af0 = mkfrag(a0p[2 * hv], a0p[2 * hv + 1]);
        bf16x16 af1 = mkfrag(a1p[2 * hv], a1p[2 * hv + 1]);
        if (k0 + 32 < TC) {
            __builtin_prefetch(arow0 + k0 + 32, 0, 1);
            __builtin_prefetch(arow1 + k0 + 32, 0, 1);
        }
#pragma unroll
        for (int c = 0; c < 4; ++c) {
            const u16x8* bp = (const u16x8*)(bbase + (size_t)(16 * c) * TC + k0);
            bf16x16 bf = mkfrag(bp[0], bp[1]);
            acc[0][c] = wmma_bf16(af0, bf, acc[0][c]);
            acc[1][c] = wmma_bf16(af1, bf, acc[1][c]);
        }
    }

#pragma unroll
    for (int g = 0; g < 2; ++g)
#pragma unroll
        for (int c = 0; c < 4; ++c)
#pragma unroll
            for (int j = 0; j < 8; ++j) {
                int r   = m0 + wave * 32 + g * 16 + j + hv * 8;
                int col = n0 + 16 * c + l16;
                out[(size_t)r * TC + col] = acc[g][c][j] + bo[col];
            }
}

// ---------------------------------------------------------------------------
extern "C" void kernel_launch(void* const* d_in, const int* in_sizes, int n_in,
                              void* d_out, int out_size, void* d_ws, size_t ws_size,
                              hipStream_t stream) {
    (void)in_sizes; (void)n_in; (void)out_size; (void)ws_size;
    const float* x    = (const float*)d_in[0];
    const float* Wqkv = (const float*)d_in[1];
    const float* bqkv = (const float*)d_in[2];
    const float* Wo   = (const float*)d_in[3];
    const float* bo   = (const float*)d_in[4];
    float* out        = (float*)d_out;

    const size_t nElems = (size_t)TB * TH * TT * THS;   // 8Mi
    unsigned short* Qb  = (unsigned short*)d_ws;
    unsigned short* Kb  = Qb + nElems;
    unsigned short* Vt  = Kb + nElems;
    unsigned short* xb  = Vt + nElems;                  // reused as attn output
    unsigned short* Wqt = xb + nElems;                  // [3072][1024]
    unsigned short* Wot = Wqt + (size_t)3 * TC * TC;    // [1024][1024]
    unsigned short* attnout = xb;                       // alias (xb dead by then)

    const int n4 = (TB * TT * TC) / 4;
    cvt_bf16_kernel<<<(n4 + 255) / 256, 256, 0, stream>>>(x, xb, n4);
    transpose_bf16_kernel<<<dim3((3 * TC) / 32, TC / 32), 256, 0, stream>>>(
        Wqkv, Wqt, TC, 3 * TC);
    transpose_bf16_kernel<<<dim3(TC / 32, TC / 32), 256, 0, stream>>>(
        Wo, Wot, TC, TC);
    qkv_rope_kernel<<<dim3((TB * TT) / 128, (3 * TC) / 64), 128, 0, stream>>>(
        xb, Wqt, bqkv, Qb, Kb, Vt);
    attn_kernel<<<dim3(TB * TH, TT / 64), 128, 0, stream>>>(Qb, Kb, Vt, attnout);
    oproj_kernel<<<dim3((TB * TT) / 128, TC / 64), 128, 0, stream>>>(
        attnout, Wot, bo, out);
}
